// SimpleGCN_22986664968669
// MI455X (gfx1250) — matrix operand order repaired and checked by reading, not compiled, not run on hardware
//
#include <hip/hip_runtime.h>
#include <hip/hip_bf16.h>

// ---------------------------------------------------------------------------
// SimpleGCN forward for MI455X (gfx1250, wave32).
//  - bf16 WMMA (v_wmma_f32_16x16x32_bf16) for all three feature transforms
//  - activations AND weights pre-converted to bf16 -> GEMM inner loop is
//    4x global_load_b128 + 1x v_wmma (software-pipelined, no conversion VALU)
//  - aggregate-first on layer 3 (32-wide scatter instead of 385-wide)
//  - edge scatter via f32 global atomics (resolve in 192MB L2)
// ---------------------------------------------------------------------------

typedef __attribute__((ext_vector_type(16))) __bf16 v16bf;
typedef __attribute__((ext_vector_type(8)))  float  v8f;

union Frag16 { v16bf v; uint4 q[2]; };

__device__ __forceinline__ unsigned short f32_to_bf16_rne(float f) {
  union { float f; unsigned int u; } x; x.f = f;
  unsigned int u = x.u;
  unsigned int r = u + 0x7FFFu + ((u >> 16) & 1u);
  return (unsigned short)(r >> 16);
}

// ------------------------- elementwise helpers -----------------------------

__global__ void zero_kernel(float* __restrict__ p, int n) {
  int t = blockIdx.x * blockDim.x + threadIdx.x;
  if (t < n) p[t] = 0.0f;
}

// vectorized f32 -> bf16 copy (n must be a multiple of 4)
__global__ void cvt_bf16_kernel(const float* __restrict__ in, unsigned short* __restrict__ out, int n4) {
  int t = blockIdx.x * blockDim.x + threadIdx.x;
  if (t >= n4) return;
  float4 v = ((const float4*)in)[t];
  ushort4 o;
  o.x = f32_to_bf16_rne(v.x);
  o.y = f32_to_bf16_rne(v.y);
  o.z = f32_to_bf16_rne(v.z);
  o.w = f32_to_bf16_rne(v.w);
  ((ushort4*)out)[t] = o;
}

__global__ void degree_kernel(const int* __restrict__ dst, float* __restrict__ cnt, int E) {
  int e = blockIdx.x * blockDim.x + threadIdx.x;
  if (e < E) atomicAdd(&cnt[dst[e]], 1.0f);
}

__global__ void dinv_kernel(const float* __restrict__ cnt, float* __restrict__ dinv, int n) {
  int i = blockIdx.x * blockDim.x + threadIdx.x;
  if (i < n) dinv[i] = rsqrtf(cnt[i] + 1.0f);   // +1 = self loop; always > 0
}

__global__ void norm_kernel(const int* __restrict__ src, const int* __restrict__ dst,
                            const float* __restrict__ dinv, float* __restrict__ nrm, int E) {
  int e = blockIdx.x * blockDim.x + threadIdx.x;
  if (e < E) nrm[e] = dinv[src[e]] * dinv[dst[e]];
}

// acc[i,f] = (relu?)(acc[i,f] + dinv[i]^2 * m[i,f] (+ bias[f]))
// optionally also emits a bf16 copy (next layer's WMMA A operand).
__global__ void epilogue_kernel(float* __restrict__ acc, const float* __restrict__ m,
                                const float* __restrict__ dinv, const float* __restrict__ bias,
                                unsigned short* __restrict__ out_bf16,
                                int total, int F, int do_relu) {
  int t = blockIdx.x * blockDim.x + threadIdx.x;
  if (t >= total) return;
  int i = t / F, f = t % F;
  float di = dinv[i];
  float v = acc[t] + di * di * m[t];
  if (bias) v += bias[f];
  if (do_relu) v = fmaxf(v, 0.0f);
  acc[t] = v;
  if (out_bf16) out_bf16[t] = f32_to_bf16_rne(v);
}

// weight converter: W[K][N] f32  ->  Wt[Npad][K] bf16 (transposed, zero-padded)
__global__ void convert_weight_kernel(const float* __restrict__ W, unsigned short* __restrict__ Wt,
                                      int K, int N, int Npad) {
  int t = blockIdx.x * blockDim.x + threadIdx.x;
  int total = Npad * K;
  if (t >= total) return;
  int col = t / K;          // output row = original B column
  int k   = t % K;
  float v = (col < N) ? W[(size_t)k * N + col] : 0.0f;
  Wt[t] = f32_to_bf16_rne(v);
}

// ----------------------------- edge scatter --------------------------------
// F/4 threads per edge (float4 gather); consecutive lanes cover consecutive
// feature chunks of the same row -> coalesced gather of m[src].
__global__ void scatter_kernel(const float* __restrict__ m, const int* __restrict__ src,
                               const int* __restrict__ dst, const float* __restrict__ nrm,
                               float* __restrict__ agg, int E, int F) {
  int gpe = F >> 2;                         // float4 groups per edge
  int t = blockIdx.x * blockDim.x + threadIdx.x;
  int total = E * gpe;
  if (t >= total) return;
  int e = t / gpe;
  int c = t % gpe;
  int s = src[e], d = dst[e];
  float w = nrm[e];
  const float4 mv = ((const float4*)(m + (size_t)s * F))[c];
  float* out = agg + (size_t)d * F + (size_t)c * 4;
  atomicAdd(out + 0, w * mv.x);
  atomicAdd(out + 1, w * mv.y);
  atomicAdd(out + 2, w * mv.z);
  atomicAdd(out + 3, w * mv.w);
}

// ----------------------------- WMMA GEMM -----------------------------------
// C[M,N] = Ab * Bt^T (+bias) (+sigmoid on col 0), all-bf16 operands.
//   Ab : bf16 row-major [M][K], K % 32 == 0
//   Bt : bf16 [Npad][K]  (B transposed, cols padded with zeros, Npad%16==0)
// One wave per 16x16 output tile.  Fragments per ISA 7.12.2 (lane = l + 16h):
//   A elem e -> K = k0 + 8h + e (e<8), k0 + 16 + 8h + (e-8) : 2x b128
//   B elem e -> K = k0 + 8h + e (e<8), k0 + 16 + 8h + (e-8) : 2x b128
//   C/D: acc[r] = C[tm*16 + r + 8h][tn*16 + l]
// K loop is 2-deep software pipelined (loads of slab k+1 issue before the
// WMMA of slab k).  All branches in the loop are wave-uniform -> EXEC all-1.
__global__ void gemm_bf16_wmma(const unsigned short* __restrict__ Ab,
                               const unsigned short* __restrict__ Bt,
                               const float* __restrict__ bias, float* __restrict__ C,
                               int M, int K, int N, int sigmoid_col0) {
  int wave = (blockIdx.x * blockDim.x + threadIdx.x) >> 5;
  int lane = threadIdx.x & 31;
  int tn_cnt = (N + 15) >> 4;
  int tm_cnt = (M + 15) >> 4;
  if (wave >= tm_cnt * tn_cnt) return;      // wave-uniform exit
  int tm = wave / tn_cnt;
  int tn = wave % tn_cnt;
  int h = lane >> 4;
  int l = lane & 15;
  int arow = tm * 16 + l;
  if (arow >= M) arow = M - 1;              // clamp (only matters if M%16!=0)
  int bcol = tn * 16 + l;                   // always < Npad: load is safe

  const unsigned short* ap = Ab + (size_t)arow * K + 8 * h;
  const unsigned short* bp = Bt + (size_t)bcol * K + 8 * h;

  Frag16 a0, b0, a1, b1;
  a0.q[0] = *(const uint4*)(ap);
  a0.q[1] = *(const uint4*)(ap + 16);
  b0.q[0] = *(const uint4*)(bp);
  b0.q[1] = *(const uint4*)(bp + 16);
  a1 = a0; b1 = b0;

  v8f acc = {};
  for (int k0 = 0; k0 < K; k0 += 32) {
    int k1 = k0 + 32;
    if (k1 < K) {                           // uniform: prefetch next slab
      a1.q[0] = *(const uint4*)(ap + k1);
      a1.q[1] = *(const uint4*)(ap + k1 + 16);
      b1.q[0] = *(const uint4*)(bp + k1);
      b1.q[1] = *(const uint4*)(bp + k1 + 16);
    }
    acc = __builtin_amdgcn_wmma_f32_16x16x32_bf16(
        /*neg_a=*/false, a0.v, /*neg_b=*/false, b0.v,
        /*c_mod=*/(short)0, acc, /*reuse_a=*/false, /*reuse_b=*/false);
    a0 = a1; b0 = b1;
  }

  // epilogue: n is per-lane constant across the 8 output rows
  int n = tn * 16 + l;
  bool nok = (n < N);
  float bn = 0.0f;
  if (bias && nok) bn = bias[n];
  bool sig = sigmoid_col0 && (n == 0);
#pragma unroll
  for (int r = 0; r < 8; ++r) {
    int m = tm * 16 + r + 8 * h;
    if (nok && m < M) {
      float v = acc[r] + bn;
      if (sig) v = __builtin_amdgcn_rcpf(1.0f + __expf(-v));
      C[(size_t)m * N + n] = v;
    }
  }
}

// ------------------------------- launcher ----------------------------------

static inline unsigned cdiv_u(long long a, long long b) { return (unsigned)((a + b - 1) / b); }

extern "C" void kernel_launch(void* const* d_in, const int* in_sizes, int n_in,
                              void* d_out, int out_size, void* d_ws, size_t ws_size,
                              hipStream_t stream) {
  const int F_IN = 384, F1 = 64, F2 = 32, F3 = 385;
  const int F3P = 400;                  // F3 padded to multiple of 16
  const float* x  = (const float*)d_in[0];
  const int* ei   = (const int*)d_in[1];
  const float* W1 = (const float*)d_in[2];
  const float* b1 = (const float*)d_in[3];
  const float* W2 = (const float*)d_in[4];
  const float* b2 = (const float*)d_in[5];
  const float* W3 = (const float*)d_in[6];
  const float* b3 = (const float*)d_in[7];
  float* out = (float*)d_out;

  const int Nn = in_sizes[0] / F_IN;   // 50000
  const int E  = in_sizes[1] / 2;      // 1600000
  const int* srcp = ei;
  const int* dstp = ei + E;

  // workspace carve-up: f32 buffers, then bf16 buffers (~100 MB total)
  float* ws = (float*)d_ws;
  size_t o = 0;
  float* cnt  = ws + o; o += (size_t)Nn;
  float* dinv = ws + o; o += (size_t)Nn;
  float* nrm  = ws + o; o += (size_t)E;
  float* m1   = ws + o; o += (size_t)Nn * F1;
  float* h1   = ws + o; o += (size_t)Nn * F1;   // agg1 -> h1 in place
  float* m2   = ws + o; o += (size_t)Nn * F2;
  float* h2   = ws + o; o += (size_t)Nn * F2;   // agg2 -> h2 in place
  float* agg3 = ws + o; o += (size_t)Nn * F2;
  unsigned short* xb    = (unsigned short*)(ws + o);    // bf16 x     [Nn][384]
  unsigned short* h1b   = xb  + (size_t)Nn * F_IN;      // bf16 h1    [Nn][64]
  unsigned short* agg3b = h1b + (size_t)Nn * F1;        // bf16 agg3  [Nn][32]
  unsigned short* wt1   = agg3b + (size_t)Nn * F2;      // [64][384]
  unsigned short* wt2   = wt1 + (size_t)F1 * F_IN;      // [32][64]
  unsigned short* wt3   = wt2 + (size_t)F2 * F1;        // [400][32]
  (void)ws_size; (void)n_in; (void)out_size;

  const int B = 256;

  // input + weight conversion (bf16; weights transposed + padded)
  cvt_bf16_kernel<<<cdiv_u((long long)Nn * F_IN / 4, B), B, 0, stream>>>(x, xb, Nn * F_IN / 4);
  convert_weight_kernel<<<cdiv_u((long long)F1 * F_IN, B), B, 0, stream>>>(W1, wt1, F_IN, F1, F1);
  convert_weight_kernel<<<cdiv_u((long long)F2 * F1, B), B, 0, stream>>>(W2, wt2, F1, F2, F2);
  convert_weight_kernel<<<cdiv_u((long long)F3P * F2, B), B, 0, stream>>>(W3, wt3, F2, F3, F3P);

  // zero accumulators
  zero_kernel<<<cdiv_u(Nn, B), B, 0, stream>>>(cnt, Nn);
  zero_kernel<<<cdiv_u((long long)Nn * F1, B), B, 0, stream>>>(h1, Nn * F1);
  zero_kernel<<<cdiv_u((long long)Nn * F2, B), B, 0, stream>>>(h2, Nn * F2);
  zero_kernel<<<cdiv_u((long long)Nn * F2, B), B, 0, stream>>>(agg3, Nn * F2);

  // normalization
  degree_kernel<<<cdiv_u(E, B), B, 0, stream>>>(dstp, cnt, E);
  dinv_kernel<<<cdiv_u(Nn, B), B, 0, stream>>>(cnt, dinv, Nn);
  norm_kernel<<<cdiv_u(E, B), B, 0, stream>>>(srcp, dstp, dinv, nrm, E);

  const int WPB = B / 32;  // waves per block

  // layer 1: transform-first (384 -> 64), scatter 64-wide
  {
    long long tiles = (long long)((Nn + 15) / 16) * (F1 / 16);
    gemm_bf16_wmma<<<cdiv_u(tiles, WPB), B, 0, stream>>>(xb, wt1, nullptr, m1, Nn, F_IN, F1, 0);
    scatter_kernel<<<cdiv_u((long long)E * (F1 / 4), B), B, 0, stream>>>(m1, srcp, dstp, nrm, h1, E, F1);
    epilogue_kernel<<<cdiv_u((long long)Nn * F1, B), B, 0, stream>>>(h1, m1, dinv, b1, h1b, Nn * F1, F1, 1);
  }

  // layer 2: transform-first (64 -> 32), scatter 32-wide
  {
    long long tiles = (long long)((Nn + 15) / 16) * (F2 / 16);
    gemm_bf16_wmma<<<cdiv_u(tiles, WPB), B, 0, stream>>>(h1b, wt2, nullptr, m2, Nn, F1, F2, 0);
    scatter_kernel<<<cdiv_u((long long)E * (F2 / 4), B), B, 0, stream>>>(m2, srcp, dstp, nrm, h2, E, F2);
    epilogue_kernel<<<cdiv_u((long long)Nn * F2, B), B, 0, stream>>>(h2, m2, dinv, b2, nullptr, Nn * F2, F2, 1);
  }

  // layer 3: AGGREGATE-FIRST (scatter 32-wide, then 32 -> 385 transform).
  {
    scatter_kernel<<<cdiv_u((long long)E * (F2 / 4), B), B, 0, stream>>>(h2, srcp, dstp, nrm, agg3, E, F2);
    // self-loop contribution: agg3 += dinv^2 * h2; emit bf16 for GEMM3's A
    epilogue_kernel<<<cdiv_u((long long)Nn * F2, B), B, 0, stream>>>(agg3, h2, dinv, nullptr, agg3b, Nn * F2, F2, 0);
    long long tiles = (long long)((Nn + 15) / 16) * (F3P / 16);
    gemm_bf16_wmma<<<cdiv_u(tiles, WPB), B, 0, stream>>>(agg3b, wt3, b3, out, Nn, F2, F3, 1);
  }
}
